// GCN_43155831390824
// MI455X (gfx1250) — matrix operand order repaired and checked by reading, not compiled, not run on hardware
//
#include <hip/hip_runtime.h>
#include <hip/hip_bf16.h>

#define N_NODES 10000
#define E_EDGES 320000
#define IN_F    10000
#define HID     128
#define NCLS    2
#define IN_PAD  10016   // 313 * 32
#define KCHUNKS 313

typedef __attribute__((ext_vector_type(16))) __bf16 v16bf;
typedef __attribute__((ext_vector_type(8)))  float  v8f;

static __device__ __forceinline__ __bf16 f2bf(float f) { return (__bf16)f; }

// ---------------------------------------------------------------------------
// Async global->LDS copy (CDNA5 GLOBAL_LOAD_ASYNC_TO_LDS_B128, ASYNCcnt).
// ---------------------------------------------------------------------------
static __device__ __forceinline__ void async_copy16(uint32_t lds_off, const void* gaddr) {
    asm volatile("global_load_async_to_lds_b128 %0, %1, off"
                 :: "v"(lds_off), "v"(gaddr) : "memory");
}

// ---------------------------------------------------------------------------
// Pack W_conv [IN_F x HID] (row-major f32) into Wt [HID x IN_PAD] bf16,
// transposed + zero-padded along K so B fragments are contiguous.
// ---------------------------------------------------------------------------
__global__ void k_pack_w(const float* __restrict__ W, __bf16* __restrict__ Wt) {
    int idx = blockIdx.x * 256 + threadIdx.x;          // over HID * IN_PAD
    if (idx >= HID * IN_PAD) return;
    int n = idx / IN_PAD;
    int k = idx - n * IN_PAD;
    float v = (k < IN_F) ? W[(size_t)k * HID + n] : 0.0f;
    Wt[(size_t)n * IN_PAD + k] = f2bf(v);
}

// ---------------------------------------------------------------------------
// Degree: deg[i] = 1 (self loop) + sum of ew over destination index (col).
// ---------------------------------------------------------------------------
__global__ void k_deg_init(float* __restrict__ deg) {
    int i = blockIdx.x * 256 + threadIdx.x;
    if (i < N_NODES) deg[i] = 1.0f;
}

__global__ void k_deg_edges(const int* __restrict__ ei,
                            const float* __restrict__ ew,
                            float* __restrict__ deg) {
    int e = blockIdx.x * 256 + threadIdx.x;
    if (e >= E_EDGES) return;
    atomicAdd(&deg[ei[E_EDGES + e]], ew[e]);
}

__global__ void k_dinv(float* __restrict__ deg) {
    int i = blockIdx.x * 256 + threadIdx.x;
    if (i >= N_NODES) return;
    float d = deg[i];
    deg[i] = (d > 0.0f) ? rsqrtf(d) : 0.0f;
}

// ---------------------------------------------------------------------------
// A fragment build: 16x32 bf16, contiguous k = 16*half + s per lane, f32->bf16 converted in registers.
// ---------------------------------------------------------------------------
static __device__ __forceinline__ v16bf make_afrag(const float* __restrict__ xrow,
                                                   int k0, int half) {
    v16bf a;
    const int ka = k0 + half * 16;
    if (ka < IN_F) {                                   // IN_F % 8 == 0 -> group granular
        float4 p = *(const float4*)(xrow + ka);
        float4 q = *(const float4*)(xrow + ka + 4);
        a[0] = f2bf(p.x); a[1] = f2bf(p.y); a[2] = f2bf(p.z); a[3] = f2bf(p.w);
        a[4] = f2bf(q.x); a[5] = f2bf(q.y); a[6] = f2bf(q.z); a[7] = f2bf(q.w);
    } else {
        #pragma unroll
        for (int i = 0; i < 8; ++i) a[i] = f2bf(0.0f);
    }
    const int kb = ka + 8;
    if (kb < IN_F) {
        float4 p = *(const float4*)(xrow + kb);
        float4 q = *(const float4*)(xrow + kb + 4);
        a[8]  = f2bf(p.x); a[9]  = f2bf(p.y); a[10] = f2bf(p.z); a[11] = f2bf(p.w);
        a[12] = f2bf(q.x); a[13] = f2bf(q.y); a[14] = f2bf(q.z); a[15] = f2bf(q.w);
    } else {
        #pragma unroll
        for (int i = 8; i < 16; ++i) a[i] = f2bf(0.0f);
    }
    return a;
}

// ---------------------------------------------------------------------------
// Stage one 32K x 128N bf16 B chunk (8KB) into LDS. 128 threads, 4 x 16B each.
// LDS layout: row n (0..127) at n*64, K sub-block p (0..3) at p*16.
// ---------------------------------------------------------------------------
static __device__ __forceinline__ void stage_B_chunk(const __bf16* __restrict__ Wt,
                                                     int k0, uint32_t lds_dst,
                                                     int tid) {
    const int p  = tid & 3;
    const int n0 = tid >> 2;
    const __bf16* src = Wt + (size_t)n0 * IN_PAD + k0 + p * 8;
    uint32_t dst = lds_dst + (uint32_t)(n0 * 64 + p * 16);
    #pragma unroll
    for (int j = 0; j < 4; ++j) {                      // rows n0, n0+32, n0+64, n0+96
        async_copy16(dst, (const void*)src);
        src += (size_t)32 * IN_PAD;
        dst += 32 * 64;
    }
}

// ---------------------------------------------------------------------------
// h = x @ W_conv via v_wmma_f32_16x16x32_bf16.
// Block = 4 waves; each wave computes 32 rows x 128 cols (2 row-tiles,
// 16 accumulators). B staged in LDS via async copies, triple-buffered.
// ---------------------------------------------------------------------------
__global__ void __launch_bounds__(128) k_gemm(const float* __restrict__ x,
                                              const __bf16* __restrict__ Wt,
                                              float* __restrict__ h) {
    __shared__ __align__(32) unsigned char smemB[3 * 8192];

    const int tid  = threadIdx.x;
    const int lane = tid & 31;
    const int wave = tid >> 5;
    const int half = lane >> 4;
    const int l15  = lane & 15;

    const int rowBase0 = (blockIdx.x * 8 + wave * 2) * 16;
    const int rowBase1 = rowBase0 + 16;
    const bool valid0 = rowBase0 < N_NODES;
    const bool valid1 = rowBase1 < N_NODES;
    const int r0 = valid0 ? rowBase0 : 0;              // clamp: compute garbage, skip store
    const int r1 = valid1 ? rowBase1 : 0;
    const float* xrow0 = x + (size_t)(r0 + l15) * IN_F;
    const float* xrow1 = x + (size_t)(r1 + l15) * IN_F;

    const uint32_t lds_base = (uint32_t)(uintptr_t)&smemB[0];

    v8f acc0[8] = {};
    v8f acc1[8] = {};

    // Prologue: prefetch chunks 0 and 1 into slots 0 and 1.
    stage_B_chunk(Wt, 0,  lds_base,        tid);
    stage_B_chunk(Wt, 32, lds_base + 8192, tid);

    int cur = 0;        // slot being consumed
    int nxt = 2;        // slot being filled (chunk kc+2)

    for (int kc = 0; kc < KCHUNKS; ++kc) {
        const int k0 = kc * 32;

        // Own async portions of chunk kc complete (kc+1 may stay outstanding).
        if (kc + 1 < KCHUNKS) asm volatile("s_wait_asynccnt 4" ::: "memory");
        else                  asm volatile("s_wait_asynccnt 0" ::: "memory");
        __syncthreads();   // all waves' portions of chunk kc landed; slot `nxt` free

        if (kc + 2 < KCHUNKS)
            stage_B_chunk(Wt, (kc + 2) * 32, lds_base + (uint32_t)nxt * 8192, tid);

        v16bf a0 = make_afrag(xrow0, k0, half);
        v16bf a1 = make_afrag(xrow1, k0, half);

        const unsigned char* bufp = &smemB[cur * 8192 + l15 * 64 + half * 32];
        #pragma unroll
        for (int t = 0; t < 8; ++t) {
            v16bf b = *(const v16bf*)(bufp + t * 1024);  // 2x ds_load_b128
            acc0[t] = __builtin_amdgcn_wmma_f32_16x16x32_bf16(
                false, a0, false, b, (short)0, acc0[t], false, false);
            acc1[t] = __builtin_amdgcn_wmma_f32_16x16x32_bf16(
                false, a1, false, b, (short)0, acc1[t], false, false);
        }

        cur = (cur == 2) ? 0 : cur + 1;
        nxt = (nxt == 2) ? 0 : nxt + 1;
    }

    // Store: VGPR r -> row half*8 + r, lane&15 -> col within 16-col tile.
    const int m = half * 8;
    if (valid0) {
        #pragma unroll
        for (int r = 0; r < 8; ++r) {
            float* hp = h + (size_t)(rowBase0 + m + r) * HID + l15;
            #pragma unroll
            for (int t = 0; t < 8; ++t) hp[t * 16] = acc0[t][r];
        }
    }
    if (valid1) {
        #pragma unroll
        for (int r = 0; r < 8; ++r) {
            float* hp = h + (size_t)(rowBase1 + m + r) * HID + l15;
            #pragma unroll
            for (int t = 0; t < 8; ++t) hp[t * 16] = acc1[t][r];
        }
    }
}

// ---------------------------------------------------------------------------
// Aggregation init: out[i][c] = b_conv[c] + dinv[i]^2 * h[i][c]  (self loop)
// ---------------------------------------------------------------------------
__global__ void k_agg_init(const float* __restrict__ dinv,
                           const float* __restrict__ h,
                           const float* __restrict__ bconv,
                           float* __restrict__ agg) {
    int idx = blockIdx.x * 256 + threadIdx.x;          // over N*HID
    if (idx >= N_NODES * HID) return;
    int i = idx >> 7;
    int c = idx & 127;
    float di = dinv[i];
    agg[idx] = bconv[c] + di * di * h[idx];
}

// ---------------------------------------------------------------------------
// Edge scatter: one wave per edge, lane handles 4 of 128 channels.
// ---------------------------------------------------------------------------
__global__ void __launch_bounds__(256) k_agg_edges(const int* __restrict__ ei,
                                                   const float* __restrict__ ew,
                                                   const float* __restrict__ dinv,
                                                   const float* __restrict__ h,
                                                   float* __restrict__ agg) {
    int e = blockIdx.x * 8 + (threadIdx.x >> 5);
    if (e >= E_EDGES) return;
    int lane = threadIdx.x & 31;
    int r = ei[e];
    int c = ei[E_EDGES + e];
    float norm = dinv[r] * ew[e] * dinv[c];
    float4 hv = *(const float4*)(h + (size_t)r * HID + lane * 4);
    float* op = agg + (size_t)c * HID + lane * 4;
    atomicAdd(op + 0, norm * hv.x);
    atomicAdd(op + 1, norm * hv.y);
    atomicAdd(op + 2, norm * hv.z);
    atomicAdd(op + 3, norm * hv.w);
}

// ---------------------------------------------------------------------------
// Head: ReLU -> [128 x 2] linear -> 2-class softmax. One thread per node.
// ---------------------------------------------------------------------------
__global__ void k_head(const float* __restrict__ agg,
                       const float* __restrict__ Wl,
                       const float* __restrict__ bl,
                       float* __restrict__ out) {
    int n = blockIdx.x * 256 + threadIdx.x;
    if (n >= N_NODES) return;
    float l0 = bl[0], l1 = bl[1];
    const float* row = agg + (size_t)n * HID;
    #pragma unroll 4
    for (int c = 0; c < HID; ++c) {
        float v = fmaxf(row[c], 0.0f);
        l0 += v * Wl[c * 2 + 0];
        l1 += v * Wl[c * 2 + 1];
    }
    float m  = fmaxf(l0, l1);
    float e0 = __expf(l0 - m);
    float e1 = __expf(l1 - m);
    float s  = e0 + e1;
    out[n * 2 + 0] = e0 / s;
    out[n * 2 + 1] = e1 / s;
}

// ---------------------------------------------------------------------------
extern "C" void kernel_launch(void* const* d_in, const int* in_sizes, int n_in,
                              void* d_out, int out_size, void* d_ws, size_t ws_size,
                              hipStream_t stream) {
    const float* x      = (const float*)d_in[0];   // [N, IN]
    const int*   ei     = (const int*)  d_in[1];   // [2, E]
    const float* ew     = (const float*)d_in[2];   // [E]
    const float* W_conv = (const float*)d_in[3];   // [IN, HID]
    const float* b_conv = (const float*)d_in[4];   // [HID]
    const float* W_lin  = (const float*)d_in[5];   // [HID, 2]
    const float* b_lin  = (const float*)d_in[6];   // [2]
    float* out = (float*)d_out;

    // Workspace layout (all 256B aligned):
    char* ws = (char*)d_ws;
    __bf16* Wt  = (__bf16*)(ws);                       // 128*10016*2   = 2,564,096 B
    float* h    = (float*)(ws + 2564096);              // 10000*128*4   = 5,120,000 B
    float* dinv = (float*)(ws + 7684096);              //                  40,960 B (padded)
    float* agg  = (float*)(ws + 7725056);              // 10000*128*4   = 5,120,000 B

    k_pack_w<<<(HID * IN_PAD) / 256, 256, 0, stream>>>(W_conv, Wt);
    k_deg_init <<<(N_NODES + 255) / 256, 256, 0, stream>>>(dinv);
    k_deg_edges<<<(E_EDGES + 255) / 256, 256, 0, stream>>>(ei, ew, dinv);
    k_dinv     <<<(N_NODES + 255) / 256, 256, 0, stream>>>(dinv);
    // GEMM: 625 row-tiles, 8 per block (4 waves x 2) -> 79 blocks
    k_gemm<<<(N_NODES / 16 + 7) / 8, 128, 0, stream>>>(x, Wt, h);
    k_agg_init <<<(N_NODES * HID) / 256, 256, 0, stream>>>(dinv, h, b_conv, agg);
    k_agg_edges<<<E_EDGES / 8, 256, 0, stream>>>(ei, ew, dinv, h, agg);
    k_head<<<(N_NODES + 255) / 256, 256, 0, stream>>>(agg, W_lin, b_lin, out);
}
